// CustomizedEncDecMultiheadAttention_57758720196964
// MI455X (gfx1250) — matrix operand order repaired and checked by reading, not compiled, not run on hardware
//
#include <hip/hip_runtime.h>
#include <hip/hip_bf16.h>
#include <math.h>

typedef __bf16 bf16_t;
typedef __attribute__((ext_vector_type(16))) __bf16 v16bf;
typedef __attribute__((ext_vector_type(8)))  __bf16 v8bf;
typedef __attribute__((ext_vector_type(4)))  __bf16 v4bf;
typedef __attribute__((ext_vector_type(8)))  float  v8f;

// ---------------------------------------------------------------------------
// Async 16-byte global -> LDS copy (CDNA5 GLOBAL_LOAD_ASYNC_TO_LDS_B128,
// tracked by ASYNCcnt). lds_byte_off = addrspace(3) byte offset (low 32 bits
// of the flat address ARE the LDS offset per ISA 10.2 aperture mapping).
// ---------------------------------------------------------------------------
__device__ __forceinline__ void async_copy_b128(unsigned lds_byte_off,
                                                const void* gptr) {
  asm volatile("global_load_async_to_lds_b128 %0, %1, off"
               :: "v"(lds_byte_off), "v"(gptr) : "memory");
}

// ---------------------------------------------------------------------------
// f32 -> bf16 convert, 8 elements/thread (packed cvt + b128 store)
// ---------------------------------------------------------------------------
__global__ __launch_bounds__(256)
void cvt_f32_bf16x8(const float* __restrict__ in, bf16_t* __restrict__ out,
                    int n8, float scale) {
  int i = blockIdx.x * 256 + threadIdx.x;
  if (i < n8) {
    const float4* p = reinterpret_cast<const float4*>(in) + (size_t)i * 2;
    float4 f0 = p[0], f1 = p[1];
    v8bf r;
    r[0] = (bf16_t)(f0.x * scale); r[1] = (bf16_t)(f0.y * scale);
    r[2] = (bf16_t)(f0.z * scale); r[3] = (bf16_t)(f0.w * scale);
    r[4] = (bf16_t)(f1.x * scale); r[5] = (bf16_t)(f1.y * scale);
    r[6] = (bf16_t)(f1.z * scale); r[7] = (bf16_t)(f1.w * scale);
    *(reinterpret_cast<v8bf*>(out) + i) = r;
  }
}

// ---------------------------------------------------------------------------
// WMMA GEMM: C[M,N] = A[M,K](bf16) * W[N,K]^T(bf16) + bias[N]*bias_scale
// 256 threads = 8 waves (wave32). Block tile BM=128, BN=128, BK=32.
// Each wave computes a 64x32 strip: 4 M-tiles x 2 N-tiles = 8 WMMAs per
// K-step; A fragments reused across both N-tiles (1.5 ds_load_b128/WMMA).
// Double-buffered LDS staged by async global->LDS copies (ASYNCcnt).
// ---------------------------------------------------------------------------
template<bool OUT_BF16>
__global__ __launch_bounds__(256)
void gemm_bf16_wmma(const bf16_t* __restrict__ A,
                    const bf16_t* __restrict__ W,
                    const float* __restrict__ bias, float bias_scale,
                    void* __restrict__ Out,
                    int M, int N, int K)
{
  __shared__ __align__(16) bf16_t As[2][128 * 32];  // 2 x 8 KB
  __shared__ __align__(16) bf16_t Bs[2][128 * 32];  // 2 x 8 KB

  const int tid  = threadIdx.x;
  const int lane = tid & 31;
  const int wid  = tid >> 5;                 // 0..7
  const int m0   = blockIdx.y * 128;
  const int n0   = blockIdx.x * 128;

  const int wm = (wid >> 2) * 64;            // 0 or 64
  const int wn = (wid & 3) * 32;             // 0,32,64,96

  // staging chunk map: 16B chunks, chunk c -> row c>>2, col (c&3)*8 bf16
  const int r0 = tid >> 2;                   // rows 0..63  (chunk tid)
  const int r1 = r0 + 64;                    // rows 64..127 (chunk tid+256)
  const int cc = (tid & 3) * 8;

  // fragment lane mapping (ISA 16-bit A 16x32 / B 32x16 layouts)
  const int lr   = lane & 15;
  const int kb_a = (lane < 16) ? 0 : 8;      // A: K 0-7/16-23 vs 8-15/24-31
  const int kb_b = (lane < 16) ? 0 : 8;      // B: K 0-7/16-23 vs 8-15/24-31

  v8f acc00 = {}, acc10 = {}, acc20 = {}, acc30 = {};   // N-tile 0 (wn)
  v8f acc01 = {}, acc11 = {}, acc21 = {}, acc31 = {};   // N-tile 1 (wn+16)

  auto stage = [&](int kt, int buf) {
    const int k0 = kt * 32;
    async_copy_b128((unsigned)(uintptr_t)&As[buf][r0 * 32 + cc],
                    A + (size_t)(m0 + r0) * K + k0 + cc);
    async_copy_b128((unsigned)(uintptr_t)&As[buf][r1 * 32 + cc],
                    A + (size_t)(m0 + r1) * K + k0 + cc);
    async_copy_b128((unsigned)(uintptr_t)&Bs[buf][r0 * 32 + cc],
                    W + (size_t)(n0 + r0) * K + k0 + cc);
    async_copy_b128((unsigned)(uintptr_t)&Bs[buf][r1 * 32 + cc],
                    W + (size_t)(n0 + r1) * K + k0 + cc);
  };

  const int nk = K >> 5;
  stage(0, 0);

  for (int kt = 0; kt < nk; ++kt) {
    const int cur = kt & 1;
    if (kt + 1 < nk) {
      stage(kt + 1, cur ^ 1);
      // the 4 just-issued asyncs may stay outstanding; asyncs complete in
      // order, so <=4 guarantees the previous tile has landed in LDS
      asm volatile("s_wait_asynccnt 0x4" ::: "memory");
    } else {
      asm volatile("s_wait_asynccnt 0x0" ::: "memory");
    }
    __syncthreads();   // publish all waves' landed tile data

    const bf16_t* asb = As[cur];
    const bf16_t* bsb = Bs[cur];

    // two B fragments: lane holds column n = wn(+16)+lr, K kb_b..+7 and kb_b+16..+23
    v16bf b0, b1;
    {
      const bf16_t* p0 = &bsb[(wn +      lr) * 32 + kb_b];
      const bf16_t* p1 = &bsb[(wn + 16 + lr) * 32 + kb_b];
      reinterpret_cast<v8bf*>(&b0)[0] = *reinterpret_cast<const v8bf*>(p0);
      reinterpret_cast<v8bf*>(&b0)[1] = *reinterpret_cast<const v8bf*>(p0 + 16);
      reinterpret_cast<v8bf*>(&b1)[0] = *reinterpret_cast<const v8bf*>(p1);
      reinterpret_cast<v8bf*>(&b1)[1] = *reinterpret_cast<const v8bf*>(p1 + 16);
    }
    // four A fragments (shared by both N-tiles)
    v16bf a0, a1, a2, a3;
    {
      const bf16_t* p0 = &asb[(wm +  0 + lr) * 32 + kb_a];
      const bf16_t* p1 = &asb[(wm + 16 + lr) * 32 + kb_a];
      const bf16_t* p2 = &asb[(wm + 32 + lr) * 32 + kb_a];
      const bf16_t* p3 = &asb[(wm + 48 + lr) * 32 + kb_a];
      reinterpret_cast<v8bf*>(&a0)[0] = *reinterpret_cast<const v8bf*>(p0);
      reinterpret_cast<v8bf*>(&a0)[1] = *reinterpret_cast<const v8bf*>(p0 + 16);
      reinterpret_cast<v8bf*>(&a1)[0] = *reinterpret_cast<const v8bf*>(p1);
      reinterpret_cast<v8bf*>(&a1)[1] = *reinterpret_cast<const v8bf*>(p1 + 16);
      reinterpret_cast<v8bf*>(&a2)[0] = *reinterpret_cast<const v8bf*>(p2);
      reinterpret_cast<v8bf*>(&a2)[1] = *reinterpret_cast<const v8bf*>(p2 + 16);
      reinterpret_cast<v8bf*>(&a3)[0] = *reinterpret_cast<const v8bf*>(p3);
      reinterpret_cast<v8bf*>(&a3)[1] = *reinterpret_cast<const v8bf*>(p3 + 16);
    }
    acc00 = __builtin_amdgcn_wmma_f32_16x16x32_bf16(false, a0, false, b0, (short)0, acc00, false, false);
    acc10 = __builtin_amdgcn_wmma_f32_16x16x32_bf16(false, a1, false, b0, (short)0, acc10, false, false);
    acc20 = __builtin_amdgcn_wmma_f32_16x16x32_bf16(false, a2, false, b0, (short)0, acc20, false, false);
    acc30 = __builtin_amdgcn_wmma_f32_16x16x32_bf16(false, a3, false, b0, (short)0, acc30, false, false);
    acc01 = __builtin_amdgcn_wmma_f32_16x16x32_bf16(false, a0, false, b1, (short)0, acc01, false, false);
    acc11 = __builtin_amdgcn_wmma_f32_16x16x32_bf16(false, a1, false, b1, (short)0, acc11, false, false);
    acc21 = __builtin_amdgcn_wmma_f32_16x16x32_bf16(false, a2, false, b1, (short)0, acc21, false, false);
    acc31 = __builtin_amdgcn_wmma_f32_16x16x32_bf16(false, a3, false, b1, (short)0, acc31, false, false);

    __syncthreads();   // all reads of buf[cur] done before it is re-staged
  }

  // epilogue: C/D layout: lanes 0-15 -> M=j, lanes 16-31 -> M=8+j, N=lane&15
  const int   mrow0 = m0 + wm + ((lane < 16) ? 0 : 8);
  const int   ng0 = n0 + wn + lr;
  const int   ng1 = ng0 + 16;
  const float bv0 = bias ? bias[ng0] * bias_scale : 0.0f;
  const float bv1 = bias ? bias[ng1] * bias_scale : 0.0f;
#define EPILOGUE(ACC, TOFF, NG, BV)                                   \
  {                                                                   \
    int mb = mrow0 + (TOFF);                                          \
    for (int j = 0; j < 8; ++j) {                                     \
      float val = ACC[j] + (BV);                                      \
      size_t idx = (size_t)(mb + j) * N + (NG);                       \
      if (OUT_BF16) reinterpret_cast<bf16_t*>(Out)[idx] = (bf16_t)val;\
      else          reinterpret_cast<float*>(Out)[idx]  = val;        \
    }                                                                 \
  }
  EPILOGUE(acc00,  0, ng0, bv0)
  EPILOGUE(acc10, 16, ng0, bv0)
  EPILOGUE(acc20, 32, ng0, bv0)
  EPILOGUE(acc30, 48, ng0, bv0)
  EPILOGUE(acc01,  0, ng1, bv1)
  EPILOGUE(acc11, 16, ng1, bv1)
  EPILOGUE(acc21, 32, ng1, bv1)
  EPILOGUE(acc31, 48, ng1, bv1)
#undef EPILOGUE
}

// ---------------------------------------------------------------------------
// Attention core: one block per (t,b). S=32, H=16, hd=64.
// logits/softmax in LDS; emits head-mean probs to wout[B,T,S] and the
// bf16 context for the output-projection GEMM.
// ---------------------------------------------------------------------------
__global__ __launch_bounds__(256)
void attention_kernel(const float*  __restrict__ q,    // [T*B, E]
                      const bf16_t* __restrict__ kb,   // row (t*S+s)*B+b, [.,E]
                      const bf16_t* __restrict__ vb,
                      bf16_t* __restrict__ attnb,      // [T*B, E] bf16
                      float* __restrict__ wout,        // [B, T, S]
                      int T, int S, int B, int E, int H)
{
  const int tb  = blockIdx.x;          // t*B + b
  const int t   = tb / B;
  const int b   = tb - t * B;
  const int hd  = E / H;               // 64
  const int tid = threadIdx.x;

  __shared__ float sq[1024];
  __shared__ float slog[512];          // [H][32]
  __shared__ float sprob[512];

  // cache q row
  {
    const float4* qp = reinterpret_cast<const float4*>(q + (size_t)tb * E);
    *reinterpret_cast<float4*>(&sq[tid * 4]) = qp[tid];
  }
  __syncthreads();

  // logits[h][s] = q_h . k_{t,s,b,h}
  for (int p = tid; p < H * S; p += 256) {
    int h = p >> 5;
    int s = p & 31;
    size_t krow = ((size_t)(t * S + s) * B + b) * (size_t)E + (size_t)h * hd;
    float acc = 0.f;
    #pragma unroll
    for (int d = 0; d < 64; d += 8) {
      v8bf kv = *reinterpret_cast<const v8bf*>(kb + krow + d);
      const float* qv = &sq[h * hd + d];
      #pragma unroll
      for (int j = 0; j < 8; ++j) acc += (float)kv[j] * qv[j];
    }
    slog[p] = acc;
  }
  __syncthreads();

  // softmax over S per head (tiny: 16 heads x 32)
  if (tid < H) {
    const int h = tid;
    float m = -1e30f;
    for (int s = 0; s < S; ++s) m = fmaxf(m, slog[h * 32 + s]);
    float sum = 0.f;
    for (int s = 0; s < S; ++s) {
      float e = __expf(slog[h * 32 + s] - m);
      sprob[h * 32 + s] = e;
      sum += e;
    }
    float inv = 1.0f / sum;
    for (int s = 0; s < S; ++s) sprob[h * 32 + s] *= inv;
  }
  __syncthreads();

  // attn_weights output: mean over heads -> [b][t][s]
  if (tid < S) {
    float acc = 0.f;
    for (int h = 0; h < H; ++h) acc += sprob[h * 32 + tid];
    wout[((size_t)b * T + t) * S + tid] = acc * (1.0f / (float)H);
  }

  // context: attn[h*hd+d] = sum_s p[h][s] * v[s][h*hd+d]; 4 elems per thread
  {
    const int e0 = tid * 4;
    const int h  = e0 / hd;
    float a0 = 0.f, a1 = 0.f, a2 = 0.f, a3 = 0.f;
    for (int s = 0; s < S; ++s) {
      size_t vrow = ((size_t)(t * S + s) * B + b) * (size_t)E + e0;
      v4bf vv = *reinterpret_cast<const v4bf*>(vb + vrow);
      float pscale = sprob[h * 32 + s];
      a0 += pscale * (float)vv[0];
      a1 += pscale * (float)vv[1];
      a2 += pscale * (float)vv[2];
      a3 += pscale * (float)vv[3];
    }
    v4bf r;
    r[0] = (bf16_t)a0; r[1] = (bf16_t)a1; r[2] = (bf16_t)a2; r[3] = (bf16_t)a3;
    *reinterpret_cast<v4bf*>(attnb + (size_t)tb * E + e0) = r;
  }
}

// ---------------------------------------------------------------------------
extern "C" void kernel_launch(void* const* d_in, const int* in_sizes, int n_in,
                              void* d_out, int out_size, void* d_ws, size_t ws_size,
                              hipStream_t stream) {
  const int T = 64, S = 32, B = 16, E = 1024, H = 16;
  const float scaling = 0.125f;  // (E/H)^-0.5 = 64^-0.5

  const float* query = (const float*)d_in[0];
  const float* key   = (const float*)d_in[1];
  const float* Wq    = (const float*)d_in[2];
  const float* bq    = (const float*)d_in[3];
  const float* Wk    = (const float*)d_in[4];
  const float* bk    = (const float*)d_in[5];
  const float* Wv    = (const float*)d_in[6];
  const float* bv    = (const float*)d_in[7];
  const float* Wo    = (const float*)d_in[8];
  const float* bo    = (const float*)d_in[9];

  // workspace carve-out
  char* ws = (char*)d_ws;
  size_t off = 0;
  auto carve = [&](size_t bytes) -> void* {
    void* p = ws + off;
    off += (bytes + 255) & ~(size_t)255;
    return p;
  };
  bf16_t* Wq_bf  = (bf16_t*)carve((size_t)E * E * sizeof(bf16_t));
  bf16_t* Wk_bf  = (bf16_t*)carve((size_t)E * E * sizeof(bf16_t));
  bf16_t* Wv_bf  = (bf16_t*)carve((size_t)E * E * sizeof(bf16_t));
  bf16_t* Wo_bf  = (bf16_t*)carve((size_t)E * E * sizeof(bf16_t));
  bf16_t* qry_bf = (bf16_t*)carve((size_t)T * B * E * sizeof(bf16_t));
  bf16_t* key_bf = (bf16_t*)carve((size_t)T * S * B * E * sizeof(bf16_t));
  float*  qf     = (float*) carve((size_t)T * B * E * sizeof(float));
  bf16_t* kbf    = (bf16_t*)carve((size_t)T * S * B * E * sizeof(bf16_t));
  bf16_t* vbf    = (bf16_t*)carve((size_t)T * S * B * E * sizeof(bf16_t));
  bf16_t* attnbf = (bf16_t*)carve((size_t)T * B * E * sizeof(bf16_t));
  (void)ws_size;

  float* out_w = (float*)d_out + (size_t)T * B * E;        // [B,T,S]

  // 1) one-shot bf16 conversions (key converted ONCE; 64MB bf16 fits in L2
  //    so the N-block re-reads in the K/V GEMMs are L2 hits)
  {
    const int nW8 = (E * E) / 8;
    cvt_f32_bf16x8<<<(nW8 + 255) / 256, 256, 0, stream>>>(Wq, Wq_bf, nW8, scaling);
    cvt_f32_bf16x8<<<(nW8 + 255) / 256, 256, 0, stream>>>(Wk, Wk_bf, nW8, 1.0f);
    cvt_f32_bf16x8<<<(nW8 + 255) / 256, 256, 0, stream>>>(Wv, Wv_bf, nW8, 1.0f);
    cvt_f32_bf16x8<<<(nW8 + 255) / 256, 256, 0, stream>>>(Wo, Wo_bf, nW8, 1.0f);
    const int nQ8 = (T * B * E) / 8;
    cvt_f32_bf16x8<<<(nQ8 + 255) / 256, 256, 0, stream>>>(query, qry_bf, nQ8, 1.0f);
    const int nK8 = (T * S * B * E) / 8;
    cvt_f32_bf16x8<<<(nK8 + 255) / 256, 256, 0, stream>>>(key, key_bf, nK8, 1.0f);
  }

  // 2) projections (WMMA GEMMs)
  {
    dim3 gq(E / 128, (T * B) / 128);         // (8, 8)
    gemm_bf16_wmma<false><<<gq, 256, 0, stream>>>(qry_bf, Wq_bf, bq, scaling,
                                                  (void*)qf, T * B, E, E);
    dim3 gkv(E / 128, (T * S * B) / 128);    // (8, 256)
    gemm_bf16_wmma<true><<<gkv, 256, 0, stream>>>(key_bf, Wk_bf, bk, 1.0f,
                                                  (void*)kbf, T * S * B, E, E);
    gemm_bf16_wmma<true><<<gkv, 256, 0, stream>>>(key_bf, Wv_bf, bv, 1.0f,
                                                  (void*)vbf, T * S * B, E, E);
  }

  // 3) attention core (+ attn_weights output)
  attention_kernel<<<T * B, 256, 0, stream>>>(qf, kbf, vbf, attnbf, out_w,
                                              T, S, B, E, H);

  // 4) output projection straight into d_out
  {
    dim3 go(E / 128, (T * B) / 128);
    gemm_bf16_wmma<false><<<go, 256, 0, stream>>>(attnbf, Wo_bf, bo, 1.0f,
                                                  (void*)d_out, T * B, E, E);
  }
  (void)in_sizes; (void)n_in; (void)out_size;
}
